// ComprehensiveNormalization_60447369724325
// MI455X (gfx1250) — matrix operand order repaired and checked by reading, not compiled, not run on hardware
//
#include <hip/hip_runtime.h>
#include <math.h>

// ---------------------------------------------------------------------------
// ComprehensiveNormalization for MI455X (gfx1250, wave32, WMMA bf16)
//
// Shapes (hardcoded from reference): B=4, S=2048, D=1024 -> T=8192 tokens,
// 6D=6144. MEM=512, NOI=64, RES=32.  EPS=1e-5.
//
// GEMM tiling: workgroup = 32 tokens x 1024 cols, 8 waves x 128 cols,
// 2 M-tiles per wave -> inner loop: 8 B-loads feed 16 v_wmma (bf16, f32 acc).
// ---------------------------------------------------------------------------

typedef __attribute__((ext_vector_type(16))) __bf16 v16bf;
typedef __attribute__((ext_vector_type(8)))  float  v8f;

#define TOKENS     8192
#define D_MODEL    1024
#define K_CAT      6144
#define EPS_LN     1e-5f

// ---- workspace layout (bytes, 256-aligned) --------------------------------
#define WS_W1T_OFF   0u                     // 1024*6144 bf16 = 12,582,912
#define WS_W2T_OFF   12582912u              // 1024*1024 bf16 =  2,097,152
#define WS_MID_OFF   14680064u              // 8192*1024 bf16 = 16,777,216
#define WS_EFF_OFF   31457280u              // 3*4*1024  f32  =     49,152
#define WS_STAT_OFF  31506432u              // 8192*4    f32  =    131,072
// total ~31.64 MB

static __device__ __forceinline__ unsigned short f2bfbits(float f) {
    unsigned u = __builtin_bit_cast(unsigned, f);
    u += 0x7fffu + ((u >> 16) & 1u);        // round-to-nearest-even
    return (unsigned short)(u >> 16);
}

static __device__ __forceinline__ float silu_f(float z) {
    return z / (1.0f + __expf(-z));
}

// ---------------------------------------------------------------------------
// Kernel 1: transpose fp32 [K,N] -> bf16 [N,K]  (write-coalesced)
// ---------------------------------------------------------------------------
__global__ void transpose_bf16_kernel(const float* __restrict__ src,
                                      unsigned short* __restrict__ dst,
                                      int K, int N) {
    size_t idx = (size_t)blockIdx.x * 256 + threadIdx.x;
    size_t total = (size_t)K * (size_t)N;
    if (idx >= total) return;
    int n = (int)(idx / (size_t)K);
    int k = (int)(idx % (size_t)K);
    dst[idx] = f2bfbits(src[(size_t)k * N + n]);
}

// ---------------------------------------------------------------------------
// Kernel 2: per-batch state MLP + LN  (one block per batch; launched 3x)
//   eff[b][i] = LN( silu(state@W1 + b1) @ W2 + b2 , g, be )
// ---------------------------------------------------------------------------
__global__ __launch_bounds__(256) void mlp_ln_effect_kernel(
    const float* __restrict__ state, int Ks,
    const float* __restrict__ W1, const float* __restrict__ b1,
    const float* __restrict__ W2, const float* __restrict__ b2,
    const float* __restrict__ g,  const float* __restrict__ be,
    float* __restrict__ effout /* [4][1024] */) {
    __shared__ float sst[512];
    __shared__ float hmid[1024];
    __shared__ float red[16];
    __shared__ float s_mean, s_rstd;

    const int b   = blockIdx.x;
    const int tid = threadIdx.x;
    const int lane = tid & 31, wid = tid >> 5;

    for (int k = tid; k < Ks; k += 256) sst[k] = state[(size_t)b * Ks + k];
    __syncthreads();

    for (int j = tid; j < D_MODEL; j += 256) {
        float a = b1[j];
        for (int k = 0; k < Ks; ++k) a += sst[k] * W1[(size_t)k * D_MODEL + j];
        hmid[j] = silu_f(a);
    }
    __syncthreads();

    float o[4], lsum = 0.f, lsq = 0.f;
    #pragma unroll
    for (int q = 0; q < 4; ++q) {
        int i = tid + q * 256;
        float a = b2[i];
        for (int k = 0; k < D_MODEL; ++k) a += hmid[k] * W2[(size_t)k * D_MODEL + i];
        o[q] = a; lsum += a; lsq += a * a;
    }
    #pragma unroll
    for (int m = 16; m >= 1; m >>= 1) {
        lsum += __shfl_xor(lsum, m, 32);
        lsq  += __shfl_xor(lsq,  m, 32);
    }
    if (lane == 0) { red[wid] = lsum; red[8 + wid] = lsq; }
    __syncthreads();
    if (tid == 0) {
        float s = 0.f, q2 = 0.f;
        for (int w = 0; w < 8; ++w) { s += red[w]; q2 += red[8 + w]; }
        float mean = s * (1.0f / D_MODEL);
        float var  = q2 * (1.0f / D_MODEL) - mean * mean;
        s_mean = mean; s_rstd = rsqrtf(var + EPS_LN);
    }
    __syncthreads();
    #pragma unroll
    for (int q = 0; q < 4; ++q) {
        int i = tid + q * 256;
        effout[(size_t)b * D_MODEL + i] = (o[q] - s_mean) * s_rstd * g[i] + be[i];
    }
}

// ---------------------------------------------------------------------------
// Kernel 3: per-token stats: mean/rstd of x, and of h1 = xhat*pw_g+pw_b
//   one wave per token (8 tokens per 256-thread block)
// ---------------------------------------------------------------------------
__global__ __launch_bounds__(256) void token_stats_kernel(
    const float* __restrict__ x, const int* __restrict__ pid,
    const float* __restrict__ pw_g, const float* __restrict__ pw_b,
    float* __restrict__ stats /* [T][4] */) {
    const int wid = threadIdx.x >> 5, lane = threadIdx.x & 31;
    const int tok = blockIdx.x * 8 + wid;
    const float* xr = x + (size_t)tok * D_MODEL;

    float s = 0.f, sq = 0.f;
    #pragma unroll 8
    for (int q = 0; q < 32; ++q) { float v = xr[lane + 32 * q]; s += v; sq += v * v; }
    #pragma unroll
    for (int m = 16; m >= 1; m >>= 1) { s += __shfl_xor(s, m, 32); sq += __shfl_xor(sq, m, 32); }
    float mx = s * (1.0f / D_MODEL);
    float vx = sq * (1.0f / D_MODEL) - mx * mx;
    float rx = rsqrtf(vx + EPS_LN);

    const int p = pid[tok];
    const float* gr = pw_g + (size_t)p * D_MODEL;
    const float* br = pw_b + (size_t)p * D_MODEL;
    float s2 = 0.f, q2 = 0.f;
    #pragma unroll 8
    for (int q = 0; q < 32; ++q) {
        int c = lane + 32 * q;
        float h1 = (xr[c] - mx) * rx * gr[c] + br[c];
        s2 += h1; q2 += h1 * h1;
    }
    #pragma unroll
    for (int m = 16; m >= 1; m >>= 1) { s2 += __shfl_xor(s2, m, 32); q2 += __shfl_xor(q2, m, 32); }
    float mh = s2 * (1.0f / D_MODEL);
    float vh = q2 * (1.0f / D_MODEL) - mh * mh;
    if (lane == 0) {
        stats[(size_t)tok * 4 + 0] = mx;
        stats[(size_t)tok * 4 + 1] = rx;
        stats[(size_t)tok * 4 + 2] = mh;
        stats[(size_t)tok * 4 + 3] = rsqrtf(vh + EPS_LN);
    }
}

// ---------------------------------------------------------------------------
// Kernel 4: GEMM1 fused:  mid = bf16( silu( cat @ int_W1 + b1 ) )
//   block = 32 tokens x 1024 cols, 8 waves (128 cols, 2 M-tiles each).
//   cat branch tiles generated into dynamic LDS in WMMA A-fragment order.
//   Dynamic LDS: 64KB branch tile + 1KB per-row params (gfx1250: 320KB/WGP).
// ---------------------------------------------------------------------------
__global__ __launch_bounds__(256) void gemm1_cat_kernel(
    const float* __restrict__ x,
    const int* __restrict__ pid, const int* __restrict__ cid,
    const int* __restrict__ tmi, const int* __restrict__ sci,
    const float* __restrict__ pw_g, const float* __restrict__ pw_b,
    const float* __restrict__ cp_g, const float* __restrict__ cp_b,
    const float* __restrict__ tm_g, const float* __restrict__ tm_b,
    const float* __restrict__ ms_g, const float* __restrict__ ms_b,
    const float* __restrict__ eff   /* [3][4][1024] */,
    const float* __restrict__ stats /* [T][4] */,
    const float* __restrict__ aw,
    const unsigned short* __restrict__ W1t /* [1024][6144] bf16 */,
    const float* __restrict__ b1,
    unsigned short* __restrict__ mid /* [T][1024] bf16 */) {

    extern __shared__ char smem_raw[];
    unsigned short* abuf = (unsigned short*)smem_raw;            // 32*1024 halves
    float* rmx = (float*)(smem_raw + 32 * D_MODEL * 2);          // [32]
    float* rrx = rmx + 32;
    float* rmh = rrx + 32;
    float* rrh = rmh + 32;
    int*   rpi = (int*)(rrh + 32);                               // [32]
    int*   rci = rpi + 32;
    int*   rti = rci + 32;
    int*   rsi = rti + 32;

    const int tid  = threadIdx.x;
    const int lane = tid & 31, wid = tid >> 5;
    const int hi   = lane >> 4, l16 = lane & 15;
    const int t0   = blockIdx.x * 32;
    const int cb   = wid * 128;

    if (tid < 32) {
        int tok = t0 + tid;
        rmx[tid] = stats[(size_t)tok * 4 + 0];
        rrx[tid] = stats[(size_t)tok * 4 + 1];
        rmh[tid] = stats[(size_t)tok * 4 + 2];
        rrh[tid] = stats[(size_t)tok * 4 + 3];
        rpi[tid] = pid[tok]; rci[tid] = cid[tok];
        rti[tid] = tmi[tok]; rsi[tid] = sci[tok];
    }

    // softmax(aw) -> w6 (lane-uniform scalar work)
    float w6[6];
    {
        float a[6], mx = -1e30f;
        #pragma unroll
        for (int i = 0; i < 6; ++i) { a[i] = aw[i]; mx = fmaxf(mx, a[i]); }
        float s = 0.f;
        #pragma unroll
        for (int i = 0; i < 6; ++i) { a[i] = __expf(a[i] - mx); s += a[i]; }
        float inv = 1.0f / s;
        #pragma unroll
        for (int i = 0; i < 6; ++i) w6[i] = a[i] * inv;
    }
    __syncthreads();

    v8f acc0[8], acc1[8];
    #pragma unroll
    for (int nt = 0; nt < 8; ++nt) {
        acc0[nt] = (v8f){0.f,0.f,0.f,0.f,0.f,0.f,0.f,0.f};
        acc1[nt] = (v8f){0.f,0.f,0.f,0.f,0.f,0.f,0.f,0.f};
    }

    for (int br = 0; br < 6; ++br) {
        const float wbr = w6[br];
        // ---- generate branch tile (32 tokens x 1024 cols) into abuf -------
        // layout: ((chunk*2 + mtile)*32 + fragLane)*16 + half
        #pragma unroll 4
        for (int q = 0; q < 128; ++q) {
            int e     = q * 256 + tid;
            int chunk = e >> 10;
            int rem   = e & 1023;
            int mt    = rem >> 9;        // which 16-row M-tile
            int rem2  = rem & 511;
            int L     = rem2 >> 4;       // fragment "lane"
            int i     = rem2 & 15;       // fragment half index
            int row   = mt * 16 + (L & 15);
            int klocal = i + ((L >> 4) << 4);
            int c     = (chunk << 5) + klocal;        // 0..1023
            int tok   = t0 + row;
            float xv  = x[(size_t)tok * D_MODEL + c];
            float v;
            if (br == 0) {
                float xh = (xv - rmx[row]) * rrx[row];
                size_t po = (size_t)rpi[row] * D_MODEL + c;
                float h1 = xh * pw_g[po] + pw_b[po];
                size_t co = (size_t)rci[row] * D_MODEL + c;
                v = (h1 - rmh[row]) * rrh[row] * cp_g[co] + cp_b[co];
            } else if (br == 1) {
                float xh = (xv - rmx[row]) * rrx[row];
                size_t to = (size_t)rti[row] * D_MODEL + c;
                v = xh * tm_g[to] + tm_b[to];
            } else if (br == 5) {
                float xh = (xv - rmx[row]) * rrx[row];
                size_t so = (size_t)rsi[row] * D_MODEL + c;
                v = xh * ms_g[so] + ms_b[so];
            } else {
                int bidx = tok >> 11;                 // token / 2048
                int ei   = br - 2;                    // 0:mem 1:noi 2:res
                v = xv + eff[(size_t)(ei * 4 + bidx) * D_MODEL + c];
            }
            abuf[e] = f2bfbits(wbr * v);
        }
        __syncthreads();

        // ---- WMMA over this branch's 1024 K-elements ----------------------
        const int kbase = br * D_MODEL;
        for (int kk = 0; kk < 32; ++kk) {
            const int abase = (kk * 64 + lane) << 4;
            v16bf afrag0 = *reinterpret_cast<const v16bf*>(&abuf[abase]);
            v16bf afrag1 = *reinterpret_cast<const v16bf*>(&abuf[abase + 512]);
            const int kglob = kbase + kk * 32 + 16 * hi;
            __builtin_prefetch(&W1t[(size_t)(cb + l16) * K_CAT + kglob + 32], 0, 3);
            #pragma unroll
            for (int nt = 0; nt < 8; ++nt) {
                const int n = cb + nt * 16 + l16;
                v16bf bfrag = *reinterpret_cast<const v16bf*>(&W1t[(size_t)n * K_CAT + kglob]);
                acc0[nt] = __builtin_amdgcn_wmma_f32_16x16x32_bf16(
                    false, afrag0, false, bfrag, (short)0, acc0[nt], false, false);
                acc1[nt] = __builtin_amdgcn_wmma_f32_16x16x32_bf16(
                    false, afrag1, false, bfrag, (short)0, acc1[nt], false, false);
            }
        }
        __syncthreads();
    }

    // ---- epilogue: +b1, SiLU, bf16 store --------------------------------
    #pragma unroll
    for (int nt = 0; nt < 8; ++nt) {
        const int n  = cb + nt * 16 + l16;
        const float bv = b1[n];
        #pragma unroll
        for (int j = 0; j < 8; ++j) {
            const int m = j + 8 * hi;
            float z0 = acc0[nt][j] + bv;
            float z1 = acc1[nt][j] + bv;
            mid[(size_t)(t0 + m) * D_MODEL + n]      = f2bfbits(silu_f(z0));
            mid[(size_t)(t0 + 16 + m) * D_MODEL + n] = f2bfbits(silu_f(z1));
        }
    }
}

// ---------------------------------------------------------------------------
// Kernel 5: GEMM2 + fused LayerNorm:  out = LN( mid @ int_W2 + b2, g, be )
//   block = 32 tokens x 1024 cols (2 M-tiles per wave).
// ---------------------------------------------------------------------------
__global__ __launch_bounds__(256) void gemm2_ln_kernel(
    const unsigned short* __restrict__ mid /* [T][1024] bf16 */,
    const unsigned short* __restrict__ W2t /* [1024][1024] bf16 */,
    const float* __restrict__ b2,
    const float* __restrict__ g, const float* __restrict__ be,
    float* __restrict__ out /* [T][1024] f32 */) {

    __shared__ float wsum[8][32];
    __shared__ float wsq[8][32];
    __shared__ float smean[32], srstd[32];

    const int tid  = threadIdx.x;
    const int lane = tid & 31, wid = tid >> 5;
    const int hi   = lane >> 4, l16 = lane & 15;
    const int t0   = blockIdx.x * 32;
    const int cb   = wid * 128;

    v8f acc0[8], acc1[8];
    #pragma unroll
    for (int nt = 0; nt < 8; ++nt) {
        acc0[nt] = (v8f){0.f,0.f,0.f,0.f,0.f,0.f,0.f,0.f};
        acc1[nt] = (v8f){0.f,0.f,0.f,0.f,0.f,0.f,0.f,0.f};
    }

    const unsigned short* arow0 = mid + (size_t)(t0 + l16) * D_MODEL;
    const unsigned short* arow1 = mid + (size_t)(t0 + 16 + l16) * D_MODEL;
    for (int kc = 0; kc < D_MODEL; kc += 32) {
        union { struct { uint4 lo; uint4 hh; } q; v16bf v; } a0, a1;
        a0.q.lo = *reinterpret_cast<const uint4*>(arow0 + kc + 16 * hi);
        a0.q.hh = *reinterpret_cast<const uint4*>(arow0 + kc + 16 * hi + 8);
        a1.q.lo = *reinterpret_cast<const uint4*>(arow1 + kc + 16 * hi);
        a1.q.hh = *reinterpret_cast<const uint4*>(arow1 + kc + 16 * hi + 8);
        #pragma unroll
        for (int nt = 0; nt < 8; ++nt) {
            const int n = cb + nt * 16 + l16;
            v16bf bfrag = *reinterpret_cast<const v16bf*>(
                &W2t[(size_t)n * D_MODEL + kc + 16 * hi]);
            acc0[nt] = __builtin_amdgcn_wmma_f32_16x16x32_bf16(
                false, a0.v, false, bfrag, (short)0, acc0[nt], false, false);
            acc1[nt] = __builtin_amdgcn_wmma_f32_16x16x32_bf16(
                false, a1.v, false, bfrag, (short)0, acc1[nt], false, false);
        }
    }

    // z = acc + b2; per-row partial sums for LN
    float rs0[8], rq0[8], rs1[8], rq1[8];
    #pragma unroll
    for (int j = 0; j < 8; ++j) { rs0[j] = rq0[j] = rs1[j] = rq1[j] = 0.f; }
    #pragma unroll
    for (int nt = 0; nt < 8; ++nt) {
        const int n = cb + nt * 16 + l16;
        const float bv = b2[n];
        #pragma unroll
        for (int j = 0; j < 8; ++j) {
            float z0 = acc0[nt][j] + bv;
            float z1 = acc1[nt][j] + bv;
            acc0[nt][j] = z0; acc1[nt][j] = z1;
            rs0[j] += z0; rq0[j] += z0 * z0;
            rs1[j] += z1; rq1[j] += z1 * z1;
        }
    }
    // butterfly over the 16 lanes sharing each row-half
    #pragma unroll
    for (int j = 0; j < 8; ++j) {
        #pragma unroll
        for (int m = 1; m <= 8; m <<= 1) {
            rs0[j] += __shfl_xor(rs0[j], m, 16);
            rq0[j] += __shfl_xor(rq0[j], m, 16);
            rs1[j] += __shfl_xor(rs1[j], m, 16);
            rq1[j] += __shfl_xor(rq1[j], m, 16);
        }
    }
    if (l16 == 0) {
        #pragma unroll
        for (int j = 0; j < 8; ++j) {
            wsum[wid][j + 8 * hi]      = rs0[j];
            wsq [wid][j + 8 * hi]      = rq0[j];
            wsum[wid][16 + j + 8 * hi] = rs1[j];
            wsq [wid][16 + j + 8 * hi] = rq1[j];
        }
    }
    __syncthreads();
    if (tid < 32) {
        float s = 0.f, q2 = 0.f;
        #pragma unroll
        for (int w = 0; w < 8; ++w) { s += wsum[w][tid]; q2 += wsq[w][tid]; }
        float mean = s * (1.0f / D_MODEL);
        float var  = q2 * (1.0f / D_MODEL) - mean * mean;
        smean[tid] = mean; srstd[tid] = rsqrtf(var + EPS_LN);
    }
    __syncthreads();

    #pragma unroll
    for (int nt = 0; nt < 8; ++nt) {
        const int n = cb + nt * 16 + l16;
        const float gv = g[n], bev = be[n];
        #pragma unroll
        for (int j = 0; j < 8; ++j) {
            const int m = j + 8 * hi;
            out[(size_t)(t0 + m) * D_MODEL + n] =
                (acc0[nt][j] - smean[m]) * srstd[m] * gv + bev;
            out[(size_t)(t0 + 16 + m) * D_MODEL + n] =
                (acc1[nt][j] - smean[16 + m]) * srstd[16 + m] * gv + bev;
        }
    }
}

// ---------------------------------------------------------------------------
// Host launcher
// ---------------------------------------------------------------------------
extern "C" void kernel_launch(void* const* d_in, const int* in_sizes, int n_in,
                              void* d_out, int out_size, void* d_ws, size_t ws_size,
                              hipStream_t stream) {
    (void)in_sizes; (void)n_in; (void)out_size; (void)ws_size;

    const float* x     = (const float*)d_in[0];
    const int*   pid   = (const int*)  d_in[1];
    const int*   cid   = (const int*)  d_in[2];
    const int*   tmi   = (const int*)  d_in[3];
    const int*   sci   = (const int*)  d_in[4];
    const float* mem_s = (const float*)d_in[5];
    const float* noi_s = (const float*)d_in[6];
    const float* res_s = (const float*)d_in[7];
    const float* pw_g  = (const float*)d_in[8];
    const float* pw_b  = (const float*)d_in[9];
    const float* cp_g  = (const float*)d_in[10];
    const float* cp_b  = (const float*)d_in[11];
    const float* tm_g  = (const float*)d_in[12];
    const float* tm_b  = (const float*)d_in[13];
    const float* ms_g  = (const float*)d_in[14];
    const float* ms_b  = (const float*)d_in[15];
    const float* memW1 = (const float*)d_in[16];
    const float* memb1 = (const float*)d_in[17];
    const float* memW2 = (const float*)d_in[18];
    const float* memb2 = (const float*)d_in[19];
    const float* memg  = (const float*)d_in[20];
    const float* membe = (const float*)d_in[21];
    const float* noiW1 = (const float*)d_in[22];
    const float* noib1 = (const float*)d_in[23];
    const float* noiW2 = (const float*)d_in[24];
    const float* noib2 = (const float*)d_in[25];
    const float* noig  = (const float*)d_in[26];
    const float* noibe = (const float*)d_in[27];
    const float* resW1 = (const float*)d_in[28];
    const float* resb1 = (const float*)d_in[29];
    const float* resW2 = (const float*)d_in[30];
    const float* resb2 = (const float*)d_in[31];
    const float* resg  = (const float*)d_in[32];
    const float* resbe = (const float*)d_in[33];
    const float* intW1 = (const float*)d_in[34];
    const float* intb1 = (const float*)d_in[35];
    const float* intW2 = (const float*)d_in[36];
    const float* intb2 = (const float*)d_in[37];
    const float* intg  = (const float*)d_in[38];
    const float* intbe = (const float*)d_in[39];
    const float* aw    = (const float*)d_in[40];

    char* ws = (char*)d_ws;
    unsigned short* W1t   = (unsigned short*)(ws + WS_W1T_OFF);
    unsigned short* W2t   = (unsigned short*)(ws + WS_W2T_OFF);
    unsigned short* mid   = (unsigned short*)(ws + WS_MID_OFF);
    float*          eff   = (float*)(ws + WS_EFF_OFF);   // [3][4][1024]
    float*          stats = (float*)(ws + WS_STAT_OFF);  // [8192][4]

    // 1) weight transpose/convert to bf16 (N-major)
    {
        size_t tot1 = (size_t)K_CAT * D_MODEL;
        transpose_bf16_kernel<<<dim3((unsigned)((tot1 + 255) / 256)), 256, 0, stream>>>(
            intW1, W1t, K_CAT, D_MODEL);
        size_t tot2 = (size_t)D_MODEL * D_MODEL;
        transpose_bf16_kernel<<<dim3((unsigned)((tot2 + 255) / 256)), 256, 0, stream>>>(
            intW2, W2t, D_MODEL, D_MODEL);
    }

    // 2) per-batch effect vectors (mem / noi / res)
    mlp_ln_effect_kernel<<<dim3(4), 256, 0, stream>>>(
        mem_s, 512, memW1, memb1, memW2, memb2, memg, membe, eff + 0 * 4 * D_MODEL);
    mlp_ln_effect_kernel<<<dim3(4), 256, 0, stream>>>(
        noi_s,  64, noiW1, noib1, noiW2, noib2, noig, noibe, eff + 1 * 4 * D_MODEL);
    mlp_ln_effect_kernel<<<dim3(4), 256, 0, stream>>>(
        res_s,  32, resW1, resb1, resW2, resb2, resg, resbe, eff + 2 * 4 * D_MODEL);

    // 3) per-token LN statistics
    token_stats_kernel<<<dim3(TOKENS / 8), 256, 0, stream>>>(x, pid, pw_g, pw_b, stats);

    // 4) GEMM1 (cat generation fused) -> mid   [dynamic LDS: 64KB tile + 1KB params]
    {
        const size_t lds_bytes = (size_t)32 * D_MODEL * 2 + 32 * 8 * sizeof(float);
        gemm1_cat_kernel<<<dim3(TOKENS / 32), 256, lds_bytes, stream>>>(
            x, pid, cid, tmi, sci,
            pw_g, pw_b, cp_g, cp_b, tm_g, tm_b, ms_g, ms_b,
            eff, stats, aw, W1t, intb1, mid);
    }

    // 5) GEMM2 + fused LayerNorm -> out
    gemm2_ln_kernel<<<dim3(TOKENS / 32), 256, 0, stream>>>(
        mid, W2t, intb2, intg, intbe, (float*)d_out);
}
